// Conv1dBlock_79826262163575
// MI455X (gfx1250) — matrix operand order repaired and checked by reading, not compiled, not run on hardware
//
#include <hip/hip_runtime.h>
#include <hip/hip_bf16.h>

// ---------------------------------------------------------------------------
// Shapes (fixed by the reference): x:[T=4,B=64,C=256,L=512], W:[256,256,5]
// ---------------------------------------------------------------------------
#define T_STEPS 4
#define BATCH   64
#define CH      256      // Cin == Cout
#define LEN     512
#define NB      256      // T*B  (conv batch)
#define LP      516      // LEN + 2*PAD, zero halo baked into spike buffer
#define KW      5

typedef __attribute__((ext_vector_type(16))) _Float16 v16h;
typedef __attribute__((ext_vector_type(8)))  _Float16 v8h;
typedef __attribute__((ext_vector_type(8)))  float    v8f;

// ---- CDNA5 async global->LDS copy via inline asm (device pass only) -------
// global_load_async_to_lds_b128: VDST = per-lane LDS byte address (low 32
// bits of the generic pointer == LDS offset per the aperture mapping),
// VADDR = 64-bit global address.  Tracked by ASYNCcnt.
#if defined(__AMDGCN__)
#define USE_ASYNC_ASM 1
#endif

__device__ __forceinline__ void cp16_g2l(_Float16* ldst, const _Float16* gsrc) {
#if defined(USE_ASYNC_ASM)
  unsigned lds_off = (unsigned)(unsigned long long)ldst;
  asm volatile("global_load_async_to_lds_b128 %0, %1, off"
               :: "v"(lds_off), "v"(gsrc)
               : "memory");
#else
  *(v8h*)ldst = *(const v8h*)gsrc;
#endif
}

__device__ __forceinline__ void wait_async_done() {
#if defined(USE_ASYNC_ASM)
  asm volatile("s_wait_asynccnt 0x0" ::: "memory");
#endif
}

// ---------------------------------------------------------------------------
// Kernel 1: repack conv weights f32 [co][ci][k] -> f16 [k][co][ci]
// ---------------------------------------------------------------------------
__global__ __launch_bounds__(256)
void wcvt_kernel(const float* __restrict__ w, _Float16* __restrict__ wh) {
  int i  = blockIdx.x * 256 + threadIdx.x;      // 5*256*256 total
  int ci = i & 255;
  int co = (i >> 8) & 255;
  int k  = i >> 16;
  wh[i] = (_Float16)w[((size_t)co * CH + ci) * KW + k];
}

// ---------------------------------------------------------------------------
// Kernel 2: LIF (tau=2, vth=0.5, hard reset) + transpose to sp[n][lp][ci] f16
//   v <- (v + x)/2 ; s = H(v - 0.5) ; v <- v*(1-s)
// Block: (b, 32-ch tile, 32-l tile); transpose via LDS so both global
// accesses are coalesced.
// ---------------------------------------------------------------------------
__global__ __launch_bounds__(256)
void lif_kernel(const float* __restrict__ x, _Float16* __restrict__ sp) {
  __shared__ _Float16 tile[32][34];            // [l_local][c_local], pad 34
  const int tid = threadIdx.x;
  const int tx  = tid & 31;                    // l lane
  const int ty  = tid >> 5;                    // 0..7
  const int l0  = blockIdx.x * 32;
  const int c0  = blockIdx.y * 32;
  const int b   = blockIdx.z;

  float v[4] = {0.f, 0.f, 0.f, 0.f};
  for (int t = 0; t < T_STEPS; ++t) {
    const int n = t * BATCH + b;
#pragma unroll
    for (int q = 0; q < 4; ++q) {
      int c = c0 + ty + q * 8;
      float xv = x[(((size_t)n * CH + c) * LEN) + l0 + tx];
      float vv = v[q] + (xv - v[q]) * 0.5f;
      float s  = (vv - 0.5f) >= 0.f ? 1.f : 0.f;
      v[q] = vv * (1.f - s);
      tile[tx][ty + q * 8] = (_Float16)s;      // transposed store
    }
    __syncthreads();
#pragma unroll
    for (int q = 0; q < 4; ++q) {
      int lr = ty + q * 8;                     // l_local
      sp[((size_t)n * LP + (l0 + lr + 2)) * CH + (c0 + tx)] = tile[lr][tx];
    }
    __syncthreads();
  }
}

// ---------------------------------------------------------------------------
// Kernel 3: zero the +/-2 halo rows of the padded spike buffer
// ---------------------------------------------------------------------------
__global__ __launch_bounds__(256)
void halo_kernel(_Float16* __restrict__ sp) {
  int i  = blockIdx.x * 256 + threadIdx.x;     // NB*4*CH total
  int ci = i & 255;
  int r  = (i >> 8) & 3;
  int n  = i >> 10;
  int lp = (r < 2) ? r : (512 + r);            // {0,1,514,515}
  sp[((size_t)n * LP + lp) * CH + ci] = (_Float16)0.f;
}

// ---------------------------------------------------------------------------
// Kernel 4: conv1d(K=5,same) as 5 shifted WMMA GEMMs + bias.
//   y[n,co,l] = b[co] + sum_{ci,k} W[co,ci,k] * sp[n][l+k][ci]
// Workgroup = 8 waves -> 128(co) x 64(l) tile of one n.
// K loop: 8 blocks of 32 ci; spike tile (68 lp x 32 ci) double-buffered in
// LDS via async global->LDS copies; A fragments read straight from L2.
// ---------------------------------------------------------------------------
__global__ __launch_bounds__(256)
void conv_wmma_kernel(const _Float16* __restrict__ sp,
                      const _Float16* __restrict__ wh,
                      const float*    __restrict__ bias,
                      float*          __restrict__ out) {
  __shared__ _Float16 spt[2][68][40];          // 40-half row = 80B, 16B aligned
  const int tid  = threadIdx.x;
  const int lane = tid & 31;
  const int wv   = tid >> 5;                   // wave id 0..7
  const int colr = lane & 15;                  // M row (A) / N col (B,C)
  const int hi   = lane >> 4;                  // half-wave selector
  const int l0   = blockIdx.x * 64;
  const int co0  = blockIdx.y * 128;
  const int n    = blockIdx.z;

  const _Float16* spn = sp + (size_t)n * LP * CH;

  auto issueTile = [&](int buf, int ci0) {
    // 68 rows x 32 halfs = 272 x b128 chunks
    for (int i = tid; i < 68 * 4; i += 256) {
      int row = i >> 2, ch = i & 3;
      cp16_g2l(&spt[buf][row][ch * 8],
               spn + (size_t)(l0 + row) * CH + ci0 + ch * 8);
    }
  };

  const v8f vzero = {0.f, 0.f, 0.f, 0.f, 0.f, 0.f, 0.f, 0.f};
  v8f acc[4] = {vzero, vzero, vzero, vzero};

  issueTile(0, 0);
  int buf = 0;
  for (int s = 0; s < CH / 32; ++s) {
    const int ci0 = s * 32;
    wait_async_done();
    __syncthreads();
    if (s + 1 < CH / 32) issueTile(buf ^ 1, ci0 + 32);   // overlap next load

#pragma unroll
    for (int k = 0; k < KW; ++k) {
      // A fragment: 16x32 f16, lanes 0-15 hold K{0..7,16..23}, 16-31 K{8..15,24..31}
      const _Float16* ap =
          wh + ((size_t)(k * CH + co0 + wv * 16 + colr) * CH) + ci0 + hi * 8;
      v8h alo = *(const v8h*)ap;
      v8h ahi = *(const v8h*)(ap + 16);
      v16h a = __builtin_shufflevector(alo, ahi, 0,1,2,3,4,5,6,7,
                                                 8,9,10,11,12,13,14,15);
#pragma unroll
      for (int j = 0; j < 4; ++j) {
        // B fragment: lane = column l, K = ci {0..7,16..23} (lanes 0-15) / {8..15,24..31} (lanes 16-31)
        const _Float16* bp = &spt[buf][j * 16 + colr + k][hi * 8];
        v8h b0 = *(const v8h*)bp;
        v8h b1 = *(const v8h*)(bp + 16);
        v16h bfrag = __builtin_shufflevector(b0, b1, 0,1,2,3,4,5,6,7,
                                                     8,9,10,11,12,13,14,15);
        acc[j] = __builtin_amdgcn_wmma_f32_16x16x32_f16(
            false, a, false, bfrag, (short)0, acc[j], false, false);
      }
    }
    __syncthreads();
    buf ^= 1;
  }

  // Epilogue: C/D layout -> lanes 0-15: M=r, lanes 16-31: M=r+8; N=lane&15
#pragma unroll
  for (int j = 0; j < 4; ++j) {
#pragma unroll
    for (int r = 0; r < 8; ++r) {
      int co = co0 + wv * 16 + hi * 8 + r;
      int l  = l0 + j * 16 + colr;
      out[((size_t)n * CH + co) * LEN + l] = acc[j][r] + bias[co];
    }
  }
}

// ---------------------------------------------------------------------------
// Kernel 5: GroupNorm stats, one workgroup per (n, group); group block is
// contiguous (32 channels x 512).  stats = {mean, rstd} per group.
// ---------------------------------------------------------------------------
__global__ __launch_bounds__(256)
void gn_stats_kernel(const float* __restrict__ y, float* __restrict__ stats) {
  __shared__ float rs[256], rq[256];
  const int bx = blockIdx.x;                   // 0..2047
  const int n = bx >> 3, g = bx & 7;
  const int tid = threadIdx.x;
  const float* base = y + ((size_t)n * CH + g * 32) * LEN;
  float s = 0.f, q = 0.f;
  for (int i = tid; i < 32 * LEN; i += 256) {
    float v = base[i];
    s += v; q += v * v;
  }
  rs[tid] = s; rq[tid] = q;
  __syncthreads();
  for (int off = 128; off > 0; off >>= 1) {
    if (tid < off) { rs[tid] += rs[tid + off]; rq[tid] += rq[tid + off]; }
    __syncthreads();
  }
  if (tid == 0) {
    const float inv = 1.f / (32.f * LEN);
    float mean = rs[0] * inv;
    float var  = rq[0] * inv - mean * mean;
    stats[bx * 2 + 0] = mean;
    stats[bx * 2 + 1] = rsqrtf(var + 1e-5f);
  }
}

// ---------------------------------------------------------------------------
// Kernel 6: in-place normalize + affine
// ---------------------------------------------------------------------------
__global__ __launch_bounds__(256)
void gn_norm_kernel(float* __restrict__ y, const float* __restrict__ stats,
                    const float* __restrict__ gamma,
                    const float* __restrict__ beta) {
  size_t i = (size_t)blockIdx.x * 256 + threadIdx.x;
  int c = (int)((i >> 9) & 255);
  int n = (int)(i >> 17);
  int g = c >> 5;
  float mean = stats[(n * 8 + g) * 2 + 0];
  float rstd = stats[(n * 8 + g) * 2 + 1];
  y[i] = (y[i] - mean) * rstd * gamma[c] + beta[c];
}

// ---------------------------------------------------------------------------
// Launch
// ---------------------------------------------------------------------------
extern "C" void kernel_launch(void* const* d_in, const int* in_sizes, int n_in,
                              void* d_out, int out_size, void* d_ws, size_t ws_size,
                              hipStream_t stream) {
  const float* x      = (const float*)d_in[0];  // [4,64,256,512]
  const float* conv_w = (const float*)d_in[1];  // [256,256,5]
  const float* conv_b = (const float*)d_in[2];  // [256]
  const float* gamma  = (const float*)d_in[3];  // [256]
  const float* beta   = (const float*)d_in[4];  // [256]
  float* out = (float*)d_out;                   // [4,64,256,512]

  // Workspace layout (bytes): spikes f16 padded, weights f16, GN stats.
  const size_t SP_BYTES = (size_t)NB * LP * CH * sizeof(_Float16); // 67,633,152
  const size_t WH_BYTES = (size_t)KW * CH * CH * sizeof(_Float16); //    655,360
  char* ws = (char*)d_ws;
  _Float16* sp    = (_Float16*)(ws);
  _Float16* wh    = (_Float16*)(ws + SP_BYTES);
  float*    stats = (float*)   (ws + SP_BYTES + WH_BYTES);

  // 1) weight repack f32 -> f16 [k][co][ci]
  wcvt_kernel<<<(KW * CH * CH) / 256, 256, 0, stream>>>(conv_w, wh);
  // 2) LIF + transpose into padded spike buffer
  lif_kernel<<<dim3(LEN / 32, CH / 32, BATCH), 256, 0, stream>>>(x, sp);
  // 3) zero halo rows
  halo_kernel<<<(NB * 4 * CH) / 256, 256, 0, stream>>>(sp);
  // 4) WMMA conv + bias -> d_out
  conv_wmma_kernel<<<dim3(LEN / 64, CH / 128, NB), 256, 0, stream>>>(
      sp, wh, conv_b, out);
  // 5) GroupNorm stats
  gn_stats_kernel<<<NB * 8, 256, 0, stream>>>(out, stats);
  // 6) normalize in place
  gn_norm_kernel<<<(int)(((size_t)NB * CH * LEN) / 256), 256, 0, stream>>>(
      out, stats, gamma, beta);
}
